// _CodeBook_62440234549834
// MI455X (gfx1250) — matrix operand order repaired and checked by reading, not compiled, not run on hardware
//
#include <hip/hip_runtime.h>
#include <stdint.h>

typedef __bf16        v16bf  __attribute__((ext_vector_type(16)));
typedef float         v8f    __attribute__((ext_vector_type(8)));
typedef unsigned int  u32x4  __attribute__((ext_vector_type(4)));
typedef unsigned int  u32x8  __attribute__((ext_vector_type(8)));

#define N_ROWS      16384
#define C_IN        1024
#define NKEYS       256
#define C_OUT       1024
#define ROWS_PER_WG 64
#define CHUNK       32
#define NCHUNK      32   // C_IN / CHUNK

// ---------------------------------------------------------------------------
// Tensor Data Mover: build D# groups 0/1 in SGPRs and issue tensor_load_to_lds
// (2-group form: VADDR2/VADDR3 = NULL -> tensors up to 2D).  Tracked with
// TENSORcnt; completion enforced with s_wait_tensorcnt.
// ---------------------------------------------------------------------------
__device__ __forceinline__ void tdm_load_to_lds(
    uint64_t gaddr, uint32_t lds_addr, uint32_t dsize_log2,
    uint32_t tdim0, uint32_t tdim1, uint32_t tile0, uint32_t tile1,
    uint64_t stride0, uint32_t pad_int, uint32_t pad_amt, uint32_t pad_en)
{
  u32x4 g0;
  u32x8 g1;
  g0[0] = 1u;                                   // count=1 (valid), user mode
  g0[1] = lds_addr;                             // lds_addr[31:0]
  g0[2] = (uint32_t)gaddr;                      // global_addr[31:0]
  g0[3] = ((uint32_t)(gaddr >> 32) & 0x01FFFFFFu) | (2u << 30); // addr[56:32] | type=2
  g1[0] = (dsize_log2 << 16) | (pad_en << 20) | (pad_int << 22) | (pad_amt << 25);
  g1[1] = (tdim0 & 0xFFFFu) << 16;              // tensor_dim0[15:0]
  g1[2] = (tdim0 >> 16) | ((tdim1 & 0xFFFFu) << 16);
  g1[3] = (tdim1 >> 16) | (tile0 << 16);        // tile_dim0
  g1[4] = tile1 & 0xFFFFu;                      // tile_dim1 (tile_dim2 = 0)
  g1[5] = (uint32_t)stride0;                    // tensor_dim0_stride[31:0]
  g1[6] = (uint32_t)(stride0 >> 32) & 0xFFFFu;  // stride0[47:32] (stride1 = 0)
  g1[7] = 0u;
  asm volatile("tensor_load_to_lds %0, %1" :: "s"(g0), "s"(g1) : "memory");
}

// ---------------------------------------------------------------------------
// Prep: keys[256][1024] f32 -> ws:
//   [0,1KB)      : 0.5*||k||^2 per key (f32)
//   [1KB,513KB)  : keys in bf16, WMMA-B per-lane layout, chunk-major:
//       elem index = ((chunk*16 + ktile)*32 + lane)*16 + e
//       value      = keys[ktile*16 + lane%16][chunk*32 + 16*(lane/16) + e]
//   => each chunk's B operand is a contiguous 16KB block (1D TDM copy).
// ---------------------------------------------------------------------------
__global__ __launch_bounds__(256) void vq_prep(
    const float* __restrict__ keys, float* __restrict__ k2ws,
    __bf16* __restrict__ bws)
{
  const int lane = threadIdx.x & 31;
  const int w    = threadIdx.x >> 5;
  const int k    = blockIdx.x * 8 + w;            // 32 blocks * 8 waves = 256 keys
  const float4* row4 = (const float4*)(keys + (size_t)k * C_IN);

  float sum = 0.f;
  __align__(16) __bf16 tmp[32];
  #pragma unroll
  for (int j4 = 0; j4 < 8; ++j4) {                // lane owns chunk c0 = lane
    const float4 v = row4[lane * 8 + j4];
    sum += v.x * v.x + v.y * v.y + v.z * v.z + v.w * v.w;
    tmp[j4 * 4 + 0] = (__bf16)v.x;
    tmp[j4 * 4 + 1] = (__bf16)v.y;
    tmp[j4 * 4 + 2] = (__bf16)v.z;
    tmp[j4 * 4 + 3] = (__bf16)v.w;
  }
  #pragma unroll
  for (int m = 16; m >= 1; m >>= 1) sum += __shfl_xor(sum, m, 32);
  if (lane == 0) k2ws[k] = 0.5f * sum;

  const int    kt   = k >> 4;
  const int    lm   = k & 15;
  const size_t base = (size_t)(lane * 16 + kt) * 32;
  uint4* d0 = (uint4*)(bws + (base + lm) * 16);        // lane half h=0 (c offs 0..15)
  uint4* d1 = (uint4*)(bws + (base + lm + 16) * 16);   // lane half h=1 (c offs 16..31)
  const uint4* t4 = (const uint4*)tmp;
  d0[0] = t4[0]; d0[1] = t4[1];
  d1[0] = t4[2]; d1[1] = t4[3];
}

// ---------------------------------------------------------------------------
// Main: per WG of 64 rows — TDM double-buffered staging, bf16 WMMA GEMM with
// software-pipelined B-tile loads, argmin reduce, values gather.
// ---------------------------------------------------------------------------
__global__ __launch_bounds__(256) void vq_main(
    const float* __restrict__ x, const float* __restrict__ values,
    const float* __restrict__ k2ws, const __bf16* __restrict__ bws,
    float* __restrict__ out)
{
  // x chunk: 64 rows x 32 f32, TDM-padded to 36-dword rows (bank-conflict-free)
  __shared__ __align__(16) float        xbuf[2][ROWS_PER_WG * 36];
  // B chunk: 512 lane-blocks x 32B bf16, TDM-padded to 48B blocks (12 dwords)
  __shared__ __align__(16) unsigned int bbuf[2][512 * 12];
  __shared__ float k2s[NKEYS];
  __shared__ float redS[64 * 32];
  __shared__ int   redK[64 * 32];
  __shared__ int   idxA[64];

  const int tid  = threadIdx.x;
  const int lane = tid & 31;
  const int w    = tid >> 5;
  const int rg   = w & 3;        // row group (16 rows each)
  const int kh   = w >> 2;       // key half (128 keys each)
  const int h    = lane >> 4;
  const int mr   = lane & 15;

  k2s[tid] = k2ws[tid];          // 256 threads, 256 keys

  const uint64_t xg = (uint64_t)(uintptr_t)x +
                      (uint64_t)blockIdx.x * (ROWS_PER_WG * C_IN * 4);
  const uint64_t bg = (uint64_t)(uintptr_t)bws;

  const bool issuer = (__builtin_amdgcn_readfirstlane(w) == 0);
  if (issuer) {
    // x tile: 2D, data_size=4B, tile 32x64, line stride 1024, pad 4dw per 32dw
    tdm_load_to_lds(xg, (uint32_t)(uintptr_t)&xbuf[0][0], 2u,
                    C_IN, ROWS_PER_WG, CHUNK, ROWS_PER_WG,
                    (uint64_t)C_IN, 4u, 3u, 1u);
    // B tile: 1D, data_size=8B, 2048 qwords (16KB), pad 4dw per 8dw
    tdm_load_to_lds(bg, (uint32_t)(uintptr_t)&bbuf[0][0], 3u,
                    2048u, 1u, 2048u, 1u, 2048ull, 2u, 3u, 1u);
  }

  v8f acc[8];
  #pragma unroll
  for (int t = 0; t < 8; ++t) {
    #pragma unroll
    for (int v = 0; v < 8; ++v) acc[t][v] = 0.f;
  }

  const int r = rg * 16 + mr;

  for (int c = 0; c < NCHUNK; ++c) {
    __builtin_amdgcn_s_wait_tensorcnt(0);   // issuer waits chunk c; others no-op
    __syncthreads();                        // chunk c visible; c-1 consumers done
    const int buf = c & 1;
    if (issuer && (c + 1) < NCHUNK) {       // prefetch chunk c+1 during compute
      const int n  = c + 1;
      const int nb = n & 1;
      tdm_load_to_lds(xg + (uint64_t)n * (CHUNK * 4),
                      (uint32_t)(uintptr_t)&xbuf[nb][0], 2u,
                      C_IN, ROWS_PER_WG, CHUNK, ROWS_PER_WG,
                      (uint64_t)C_IN, 4u, 3u, 1u);
      tdm_load_to_lds(bg + (uint64_t)n * 16384u,
                      (uint32_t)(uintptr_t)&bbuf[nb][0], 3u,
                      2048u, 1u, 2048u, 1u, 2048ull, 2u, 3u, 1u);
    }

    // ---- issue all staging loads for this chunk up front ----
    // A operand source (contiguous K layout, matching B): half h -> K 16h..16h+15
    const float4* ax = (const float4*)&xbuf[buf][r * 36 + 16 * h];
    const float4 s0 = ax[0], s1 = ax[1], s2 = ax[2], s3 = ax[3];

    // B-tile fetch helper: two ds_load_b128 per 32x16 bf16 tile
    auto bld = [&](int t, uint4* q) {
      const uint4* bp =
          (const uint4*)&bbuf[buf][((kh * 8 + t) * 32 + lane) * 12];
      q[0] = bp[0];
      q[1] = bp[1];
    };
    uint4 qe[2], qo[2];
    bld(0, qe);                 // tile 0 in flight
    bld(1, qo);                 // tile 1 in flight

    // A conversion overlaps the in-flight B loads
    v16bf a;
    a[0]=(__bf16)s0.x;  a[1]=(__bf16)s0.y;  a[2]=(__bf16)s0.z;  a[3]=(__bf16)s0.w;
    a[4]=(__bf16)s1.x;  a[5]=(__bf16)s1.y;  a[6]=(__bf16)s1.z;  a[7]=(__bf16)s1.w;
    a[8]=(__bf16)s2.x;  a[9]=(__bf16)s2.y;  a[10]=(__bf16)s2.z; a[11]=(__bf16)s2.w;
    a[12]=(__bf16)s3.x; a[13]=(__bf16)s3.y; a[14]=(__bf16)s3.z; a[15]=(__bf16)s3.w;

    // depth-2 software pipeline: prefetch tile t+2 before WMMA t
    #pragma unroll
    for (int t = 0; t < 8; ++t) {
      uint4* cur = (t & 1) ? qo : qe;
      v16bf b;
      ((uint4*)&b)[0] = cur[0];
      ((uint4*)&b)[1] = cur[1];
      if (t + 2 < 8) bld(t + 2, cur);
      acc[t] = __builtin_amdgcn_wmma_f32_16x16x32_bf16(
          false, a, false, b, (short)0, acc[t], false, false);
    }
  }

  // ---- per-lane argmin of (0.5*||k||^2 - x.k) over this wave's 8 tiles ----
  float best[8];
  int   bidx[8];
  #pragma unroll
  for (int v = 0; v < 8; ++v) { best[v] = 3.4e38f; bidx[v] = 0; }
  #pragma unroll
  for (int t = 0; t < 8; ++t) {
    const int   k  = kh * 128 + t * 16 + mr;
    const float kk = k2s[k];
    #pragma unroll
    for (int v = 0; v < 8; ++v) {
      const float s = kk - acc[t][v];
      if (s < best[v]) { best[v] = s; bidx[v] = k; }
    }
  }
  // C/D layout: lanes 0-15 hold rows M=v, lanes 16-31 hold rows M=v+8
  #pragma unroll
  for (int v = 0; v < 8; ++v) {
    const int rr  = rg * 16 + v + 8 * h;
    const int col = mr + 16 * kh;
    redS[rr * 32 + col] = best[v];
    redK[rr * 32 + col] = bidx[v];
  }
  __syncthreads();
  if (tid < 64) {
    float bs = redS[tid * 32];
    int   bk = redK[tid * 32];
    for (int j = 1; j < 32; ++j) {
      const float s = redS[tid * 32 + j];
      if (s < bs) { bs = s; bk = redK[tid * 32 + j]; }
    }
    idxA[tid] = bk;
  }
  __syncthreads();

  // ---- gather values[idx] -> out (coalesced b128, values L2-resident) ----
  const float4* vals4 = (const float4*)values;
  float4* out4 = (float4*)(out + (size_t)blockIdx.x * (ROWS_PER_WG * C_OUT));
  for (int rr = 0; rr < ROWS_PER_WG; ++rr) {
    const int idx = idxA[rr];
    out4[rr * 256 + tid] = vals4[(size_t)idx * 256 + tid];
  }
}

// ---------------------------------------------------------------------------
extern "C" void kernel_launch(void* const* d_in, const int* in_sizes, int n_in,
                              void* d_out, int out_size, void* d_ws, size_t ws_size,
                              hipStream_t stream) {
  (void)in_sizes; (void)n_in; (void)out_size; (void)ws_size;
  const float* x      = (const float*)d_in[0];
  const float* keys   = (const float*)d_in[1];
  const float* values = (const float*)d_in[2];
  float*  out  = (float*)d_out;
  float*  k2ws = (float*)d_ws;                       // 1KB
  __bf16* bws  = (__bf16*)((char*)d_ws + 1024);      // 512KB, chunk-major B layout

  vq_prep<<<NKEYS / 8, 256, 0, stream>>>(keys, k2ws, bws);
  vq_main<<<N_ROWS / ROWS_PER_WG, 256, 0, stream>>>(x, values, k2ws, bws, out);
}
